// TransformerGAT_20263655703141
// MI455X (gfx1250) — matrix operand order repaired and checked by reading, not compiled, not run on hardware
//
// GATv2 x2 for MI455X (gfx1250): bf16 WMMA GEMMs + fp32 edge-softmax scatter.
#include <hip/hip_runtime.h>
#include <cstdint>
#include <cstddef>

#define BB 2
#define NN 8192
#define EE 131072
#define NEDGE (EE + NN)       // edges + self loops = 139264
#define FIN 128
#define HC 512                // H * channels (both layers)
#define NH 8
#define NEG_SLOPE 0.2f

typedef __attribute__((ext_vector_type(16))) __bf16 v16bf;
typedef __attribute__((ext_vector_type(8)))  float  v8f;

__device__ __forceinline__ __bf16 f2bf(float f) {
  union { float f; uint32_t u; } v; v.f = f;
  uint32_t r = (v.u + 0x7FFFu + ((v.u >> 16) & 1u)) >> 16;  // RNE
  union { uint16_t s; __bf16 b; } o; o.s = (uint16_t)r;
  return o.b;
}

__device__ __forceinline__ void atomicMaxF(float* addr, float val) {
  if (val >= 0.0f) atomicMax((int*)addr, __float_as_int(val));
  else             atomicMin((unsigned int*)addr, __float_as_uint(val));
}

// ---------------- prep kernels ----------------
__global__ void k_f32_to_bf16(const float* __restrict__ src, __bf16* __restrict__ dst, int n) {
  int i = blockIdx.x * blockDim.x + threadIdx.x;
  if (i < n) dst[i] = f2bf(src[i]);
}

// W [K x Nout] row-major f32  ->  Wt [Nout x K] row-major bf16
__global__ void k_wt(const float* __restrict__ W, __bf16* __restrict__ Wt, int K, int Nout) {
  int i = blockIdx.x * blockDim.x + threadIdx.x;
  if (i >= K * Nout) return;
  int k = i / Nout, n = i - k * Nout;
  Wt[(size_t)n * K + k] = f2bf(W[i]);
}

// ---------------- WMMA GEMM: C[MxNout] = A[MxK](bf16) * Wt[NoutxK]^T (bf16), f32 accum
// one wave computes a 16(M) x 64(N) tile = 4 accumulators
__global__ void __launch_bounds__(256)
k_gemm_bf16(const __bf16* __restrict__ A, const __bf16* __restrict__ Wt,
            float* __restrict__ C, int M, int K, int Nout) {
  const int wid  = (blockIdx.x * blockDim.x + threadIdx.x) >> 5;
  const int lane = threadIdx.x & 31;
  const int tilesN = Nout >> 6;
  if (wid >= (M >> 4) * tilesN) return;
  const int tm = (wid / tilesN) << 4;
  const int tn = (wid % tilesN) << 6;
  const int half = lane >> 4;
  const int r = lane & 15;

  v8f acc[4] = {};
  const __bf16* arow = A + (size_t)(tm + r) * K + 16 * half;   // A fragment rows
  for (int k0 = 0; k0 < K; k0 += 32) {
    union { uint4 q[2]; v16bf v; } af;
    af.q[0] = *reinterpret_cast<const uint4*>(arow + k0);        // K = k0+16h .. +7
    af.q[1] = *reinterpret_cast<const uint4*>(arow + k0 + 8);    // K = k0+16h+8 .. +15
#pragma unroll
    for (int sub = 0; sub < 4; ++sub) {
      union { uint4 q[2]; v16bf v; } bfrag;
      const __bf16* bp = Wt + (size_t)(tn + 16 * sub + r) * K + 16 * half + k0;
      bfrag.q[0] = *reinterpret_cast<const uint4*>(bp);          // K = k0+16h .. +7
      bfrag.q[1] = *reinterpret_cast<const uint4*>(bp + 8);      // K = k0+16h+8 .. +15
      acc[sub] = __builtin_amdgcn_wmma_f32_16x16x32_bf16(
          false, af.v, false, bfrag.v, (short)0, acc[sub], false, false);
    }
  }
#pragma unroll
  for (int sub = 0; sub < 4; ++sub) {
#pragma unroll
    for (int rr = 0; rr < 8; ++rr) {
      C[(size_t)(tm + rr + 8 * half) * Nout + tn + 16 * sub + r] = acc[sub][rr];
    }
  }
}

// ---------------- edge / softmax kernels ----------------
__global__ void k_init(float* __restrict__ m, float* __restrict__ den, float* __restrict__ num) {
  int i = blockIdx.x * blockDim.x + threadIdx.x;
  if (i < BB * NN * NH) { m[i] = -__builtin_inff(); den[i] = 0.0f; }
  if (i < BB * NN * HC) num[i] = 0.0f;
}

// one wave per (batch, edge): logits[e,h] = sum_c att[h,c]*leakyrelu(xl[src]+xr[dst]); atomic max per dst
__global__ void __launch_bounds__(256)
k_edge_logits(const int* __restrict__ ei, const float* __restrict__ xl,
              const float* __restrict__ xr, const float* __restrict__ att,
              float* __restrict__ logits, float* __restrict__ mmax) {
  const int gw = (blockIdx.x * blockDim.x + threadIdx.x) >> 5;
  if (gw >= BB * NEDGE) return;
  const int b = gw / NEDGE;
  const int e = gw - b * NEDGE;
  const int lane = threadIdx.x & 31;
  const int src = (e < EE) ? ei[e] : (e - EE);
  const int dst = (e < EE) ? ei[EE + e] : (e - EE);
  const int c0 = lane * 16;                       // lane covers 16 channels; 4 lanes per head
  const float4* pl = reinterpret_cast<const float4*>(xl + ((size_t)b * NN + src) * HC + c0);
  const float4* pr = reinterpret_cast<const float4*>(xr + ((size_t)b * NN + dst) * HC + c0);
  const float4* pa = reinterpret_cast<const float4*>(att + c0);
  float s = 0.0f;
#pragma unroll
  for (int q = 0; q < 4; ++q) {
    float4 l4 = pl[q], r4 = pr[q], a4 = pa[q];
    float v;
    v = l4.x + r4.x; v = (v > 0.0f) ? v : NEG_SLOPE * v; s += a4.x * v;
    v = l4.y + r4.y; v = (v > 0.0f) ? v : NEG_SLOPE * v; s += a4.y * v;
    v = l4.z + r4.z; v = (v > 0.0f) ? v : NEG_SLOPE * v; s += a4.z * v;
    v = l4.w + r4.w; v = (v > 0.0f) ? v : NEG_SLOPE * v; s += a4.w * v;
  }
  s += __shfl_xor(s, 1);
  s += __shfl_xor(s, 2);
  if ((lane & 3) == 0) {
    const int h = lane >> 2;
    logits[((size_t)b * NEDGE + e) * NH + h] = s;
    atomicMaxF(&mmax[((size_t)b * NN + dst) * NH + h], s);
  }
}

// one wave per (batch, edge): a = exp(logit - m[dst]); num[dst] += a*xl[src]; den[dst,h] += a
__global__ void __launch_bounds__(256)
k_edge_accum(const int* __restrict__ ei, const float* __restrict__ xl,
             const float* __restrict__ logits, const float* __restrict__ mmax,
             float* __restrict__ num, float* __restrict__ den) {
  const int gw = (blockIdx.x * blockDim.x + threadIdx.x) >> 5;
  if (gw >= BB * NEDGE) return;
  const int b = gw / NEDGE;
  const int e = gw - b * NEDGE;
  const int lane = threadIdx.x & 31;
  const int src = (e < EE) ? ei[e] : (e - EE);
  const int dst = (e < EE) ? ei[EE + e] : (e - EE);
  const int h = lane >> 2;
  const float lg = logits[((size_t)b * NEDGE + e) * NH + h];
  const float mm = mmax[((size_t)b * NN + dst) * NH + h];
  const float a = __expf(lg - mm);
  const int c0 = lane * 16;
  const size_t ps = ((size_t)b * NN + src) * HC + c0;
  const size_t pd = ((size_t)b * NN + dst) * HC + c0;
  const float4* pl = reinterpret_cast<const float4*>(xl + ps);
#pragma unroll
  for (int q = 0; q < 4; ++q) {
    float4 l4 = pl[q];
    __hip_atomic_fetch_add(&num[pd + 4 * q + 0], a * l4.x, __ATOMIC_RELAXED, __HIP_MEMORY_SCOPE_AGENT);
    __hip_atomic_fetch_add(&num[pd + 4 * q + 1], a * l4.y, __ATOMIC_RELAXED, __HIP_MEMORY_SCOPE_AGENT);
    __hip_atomic_fetch_add(&num[pd + 4 * q + 2], a * l4.z, __ATOMIC_RELAXED, __HIP_MEMORY_SCOPE_AGENT);
    __hip_atomic_fetch_add(&num[pd + 4 * q + 3], a * l4.w, __ATOMIC_RELAXED, __HIP_MEMORY_SCOPE_AGENT);
  }
  if ((lane & 3) == 0)
    __hip_atomic_fetch_add(&den[((size_t)b * NN + dst) * NH + h], a, __ATOMIC_RELAXED, __HIP_MEMORY_SCOPE_AGENT);
}

// normalize + bias -> bf16 (feeds layer-2 GEMM)
__global__ void k_finalize1(const float* __restrict__ num, const float* __restrict__ den,
                            const float* __restrict__ bias, __bf16* __restrict__ h1bf) {
  int i = blockIdx.x * blockDim.x + threadIdx.x;
  if (i >= BB * NN * HC) return;
  int hc = i & (HC - 1);
  int node = i >> 9;              // HC == 512
  float v = num[i] / den[(size_t)node * NH + (hc >> 6)] + bias[hc];
  h1bf[i] = f2bf(v);
}

// normalize + bias + ELU -> output f32
__global__ void k_finalize2(const float* __restrict__ num, const float* __restrict__ den,
                            const float* __restrict__ bias, float* __restrict__ out) {
  int i = blockIdx.x * blockDim.x + threadIdx.x;
  if (i >= BB * NN * HC) return;
  int hc = i & (HC - 1);
  int node = i >> 9;
  float v = num[i] / den[(size_t)node * NH + (hc >> 6)] + bias[hc];
  out[i] = (v > 0.0f) ? v : expm1f(v);
}

extern "C" void kernel_launch(void* const* d_in, const int* in_sizes, int n_in,
                              void* d_out, int out_size, void* d_ws, size_t ws_size,
                              hipStream_t stream) {
  (void)in_sizes; (void)n_in; (void)out_size; (void)ws_size;
  const float* x    = (const float*)d_in[0];
  const int*   ei   = (const int*)  d_in[1];
  const float* W1l  = (const float*)d_in[2];
  const float* W1r  = (const float*)d_in[3];
  const float* att1 = (const float*)d_in[4];
  const float* b1   = (const float*)d_in[5];
  const float* W2l  = (const float*)d_in[6];
  const float* W2r  = (const float*)d_in[7];
  const float* att2 = (const float*)d_in[8];
  const float* b2   = (const float*)d_in[9];
  float* out = (float*)d_out;

  char* ws = (char*)d_ws;
  size_t off = 0;
  auto take = [&](size_t bytes) { char* p = ws + off; off += (bytes + 255) & ~(size_t)255; return p; };

  const int M = BB * NN;                       // 16384 rows, both batches share weights
  __bf16* XBF  = (__bf16*)take((size_t)M * FIN * 2);
  __bf16* W1LT = (__bf16*)take((size_t)FIN * HC * 2);
  __bf16* W1RT = (__bf16*)take((size_t)FIN * HC * 2);
  __bf16* W2LT = (__bf16*)take((size_t)HC * HC * 2);
  __bf16* W2RT = (__bf16*)take((size_t)HC * HC * 2);
  float*  XL   = (float*) take((size_t)M * HC * 4);
  float*  XR   = (float*) take((size_t)M * HC * 4);
  float*  NUM  = (float*) take((size_t)M * HC * 4);
  float*  LOGI = (float*) take((size_t)BB * NEDGE * NH * 4);
  float*  MMAX = (float*) take((size_t)M * NH * 4);
  float*  DEN  = (float*) take((size_t)M * NH * 4);
  __bf16* H1BF = (__bf16*)take((size_t)M * HC * 2);

  // ---- prep: bf16 conversions + weight transposes
  { int n = M * FIN; k_f32_to_bf16<<<(n + 255) / 256, 256, 0, stream>>>(x, XBF, n); }
  { int n = FIN * HC;
    k_wt<<<(n + 255) / 256, 256, 0, stream>>>(W1l, W1LT, FIN, HC);
    k_wt<<<(n + 255) / 256, 256, 0, stream>>>(W1r, W1RT, FIN, HC); }
  { int n = HC * HC;
    k_wt<<<(n + 255) / 256, 256, 0, stream>>>(W2l, W2LT, HC, HC);
    k_wt<<<(n + 255) / 256, 256, 0, stream>>>(W2r, W2RT, HC, HC); }

  const int gemmBlocks = ((M / 16) * (HC / 64) * 32) / 256;   // 1024
  const int edgeBlocks = (BB * NEDGE * 32 + 255) / 256;       // 34816
  const int nodeBlocks = (BB * NN * HC + 255) / 256;          // 32768

  // ---- layer 1
  k_gemm_bf16<<<gemmBlocks, 256, 0, stream>>>(XBF, W1LT, XL, M, FIN, HC);
  k_gemm_bf16<<<gemmBlocks, 256, 0, stream>>>(XBF, W1RT, XR, M, FIN, HC);
  k_init<<<nodeBlocks, 256, 0, stream>>>(MMAX, DEN, NUM);
  k_edge_logits<<<edgeBlocks, 256, 0, stream>>>(ei, XL, XR, att1, LOGI, MMAX);
  k_edge_accum<<<edgeBlocks, 256, 0, stream>>>(ei, XL, LOGI, MMAX, NUM, DEN);
  k_finalize1<<<nodeBlocks, 256, 0, stream>>>(NUM, DEN, b1, H1BF);

  // ---- layer 2
  k_gemm_bf16<<<gemmBlocks, 256, 0, stream>>>(H1BF, W2LT, XL, M, HC, HC);
  k_gemm_bf16<<<gemmBlocks, 256, 0, stream>>>(H1BF, W2RT, XR, M, HC, HC);
  k_init<<<nodeBlocks, 256, 0, stream>>>(MMAX, DEN, NUM);
  k_edge_logits<<<edgeBlocks, 256, 0, stream>>>(ei, XL, XR, att2, LOGI, MMAX);
  k_edge_accum<<<edgeBlocks, 256, 0, stream>>>(ei, XL, LOGI, MMAX, NUM, DEN);
  k_finalize2<<<nodeBlocks, 256, 0, stream>>>(NUM, DEN, b2, out);
}
